// MultiHeadAttention_39960375722849
// MI455X (gfx1250) — matrix operand order repaired and checked by reading, not compiled, not run on hardware
//
#include <hip/hip_runtime.h>

typedef __attribute__((ext_vector_type(16))) _Float16 v16h;
typedef __attribute__((ext_vector_type(8)))  _Float16 v8h;
typedef __attribute__((ext_vector_type(8)))  float    v8f;
typedef __attribute__((ext_vector_type(4)))  float    v4f;
typedef __attribute__((ext_vector_type(4)))  int      v4i;

#define DMODEL 512
#define NHEADS 8
#define DK     64
#define SEQ    2048
#define BATCH  4
#define MTOT   (BATCH*SEQ)     // 8192

#define KP 72    // K-chunk LDS row pitch (f16):  32 rows x 64 data + pad
#define VP 40    // V-chunk LDS row pitch (f16):  64 rows x 32 data + pad
#define WP 520   // weight-slab LDS row pitch:    64 rows x 512 data + pad

union V16U { v16h v; v8h h[2]; };

static __device__ __forceinline__ v8f wmma_f16(v16h a, v16h b, v8f c) {
  return __builtin_amdgcn_wmma_f32_16x16x32_f16(false, a, false, b, (short)0, c,
                                                false, false);
}

// --- CDNA5 async global->LDS copy (16B per lane), with portable fallback ----
#if defined(__AMDGCN__) && __has_builtin(__builtin_amdgcn_global_load_async_to_lds_b128)
#define HAVE_ASYNC_LDS 1
typedef __attribute__((address_space(1))) v4i as1_v4i;   // global int4
typedef __attribute__((address_space(3))) v4i as3_v4i;   // LDS int4
static __device__ __forceinline__ void async_cp16(_Float16* lds, const _Float16* g) {
  __builtin_amdgcn_global_load_async_to_lds_b128((as1_v4i*)g, (as3_v4i*)lds, 0, 0);
}
#else
#define HAVE_ASYNC_LDS 0
static __device__ __forceinline__ void async_cp16(_Float16* lds, const _Float16* g) {
  *(v8h*)lds = *(const v8h*)g;           // sync fallback: load + ds_store
}
#endif

static __device__ __forceinline__ void wait_async0() {
#if defined(__AMDGCN__) && __has_builtin(__builtin_amdgcn_s_wait_asynccnt)
  __builtin_amdgcn_s_wait_asynccnt(0);
#elif HAVE_ASYNC_LDS
  asm volatile("s_wait_asynccnt 0x0" ::: "memory");
#endif
}

// ---------------------------------------------------------------------------
// Kernel 1: transpose four 512x512 fp32 weights -> f16 Wt[n][k] = W[k][n]
// ---------------------------------------------------------------------------
__global__ void transpose_weights_kernel(const float* __restrict__ Wq,
                                         const float* __restrict__ Wk,
                                         const float* __restrict__ Wv,
                                         const float* __restrict__ Wo,
                                         _Float16* __restrict__ Wt) {
  int idx   = blockIdx.x * blockDim.x + threadIdx.x;
  int which = idx >> 18;
  int r     = idx & ((1 << 18) - 1);
  int n     = r >> 9;
  int k     = r & 511;
  const float* W = (which == 0) ? Wq : (which == 1) ? Wk : (which == 2) ? Wv : Wo;
  Wt[idx] = (_Float16)W[k * DMODEL + n];
}

// ---------------------------------------------------------------------------
// Kernel 2: projection GEMM  Y = X[M,512](f32) @ W(f16) + bias
// Block = one 64-col n-slab (weights staged once in LDS, shared by 8 waves)
// x 8 m-tiles of 16 rows.  out_mode 0 -> [B,H,S,Dk]; 1 -> [B,H,Dk,S].
// ---------------------------------------------------------------------------
__global__ void __launch_bounds__(256)
proj_kernel(const float* __restrict__ X, const _Float16* __restrict__ Wt,
            const float* __restrict__ bias, _Float16* __restrict__ out,
            int out_mode) {
  __shared__ __align__(16) _Float16 wl[64 * WP];

  const int lane = threadIdx.x & 31;
  const int wave = threadIdx.x >> 5;
  const int nb   = blockIdx.x & 7;           // n-slab
  const int n0   = nb * 64;
  const int m0   = ((blockIdx.x >> 3) * 8 + wave) * 16;
  const int r    = lane & 15, half = lane >> 4;

  // ---- stage 64x512 f16 weight slab into LDS (4096 x 16B segments) ----
#pragma unroll
  for (int t = 0; t < 16; t++) {
    int seg = t * 256 + threadIdx.x;
    int row = seg >> 5, s8 = seg & 31;
    async_cp16(&wl[row * WP + s8 * 8], Wt + (size_t)(n0 + row) * DMODEL + s8 * 8);
  }
  wait_async0();
  __syncthreads();

  v8f acc[4] = {v8f{}, v8f{}, v8f{}, v8f{}};
  const float* arow = X + (size_t)(m0 + r) * DMODEL;

  for (int k0 = 0; k0 < DMODEL; k0 += 32) {
    v16h a;
    {
      const float* p0 = arow + k0 + half * 8;
      v4f f0 = *(const v4f*)(p0);
      v4f f1 = *(const v4f*)(p0 + 4);
      v4f f2 = *(const v4f*)(p0 + 16);
      v4f f3 = *(const v4f*)(p0 + 20);
#pragma unroll
      for (int i = 0; i < 4; i++) {
        a[i]      = (_Float16)f0[i];
        a[4 + i]  = (_Float16)f1[i];
        a[8 + i]  = (_Float16)f2[i];
        a[12 + i] = (_Float16)f3[i];
      }
    }
#pragma unroll
    for (int nf = 0; nf < 4; nf++) {
      const _Float16* bp = &wl[(nf * 16 + r) * WP + k0 + half * 8];
      V16U bf;
      bf.h[0] = *(const v8h*)bp;
      bf.h[1] = *(const v8h*)(bp + 16);
      acc[nf] = wmma_f16(a, bf.v, acc[nf]);
    }
  }

  const int b = m0 / SEQ;
#pragma unroll
  for (int nf = 0; nf < 4; nf++) {
    int   n  = n0 + nf * 16 + r;
    int   h  = n >> 6, d = n & 63;
    float bv = bias[n];
    if (out_mode == 0) {
#pragma unroll
      for (int v = 0; v < 8; v++) {
        int s = m0 + half * 8 + v - b * SEQ;
        out[(((size_t)b * NHEADS + h) * SEQ + s) * DK + d] =
            (_Float16)(acc[nf][v] + bv);
      }
    } else {
      v8h pk;
#pragma unroll
      for (int v = 0; v < 8; v++) pk[v] = (_Float16)(acc[nf][v] + bv);
      int s = m0 + half * 8 - b * SEQ;
      *(v8h*)(out + (((size_t)b * NHEADS + h) * DK + d) * SEQ + s) = pk;
    }
  }
}

// ---------------------------------------------------------------------------
// Kernel 3: flash attention.  Block = 8 waves on same (b,h), consecutive
// 16-query tiles.  K/V^T chunks (32 keys) double-buffered in LDS via async
// copies; one s_wait_asynccnt + one barrier per chunk.
// ---------------------------------------------------------------------------
__global__ void __launch_bounds__(256)
attn_kernel(const _Float16* __restrict__ qh, const _Float16* __restrict__ kh,
            const _Float16* __restrict__ vt, _Float16* __restrict__ attnH) {
  __shared__ __align__(16) _Float16 kb[2][32 * KP];
  __shared__ __align__(16) _Float16 vb[2][64 * VP];

  const int lane = threadIdx.x & 31;
  const int wave = threadIdx.x >> 5;
  const int gw   = blockIdx.x * 8 + wave;   // 4096 q-tiles
  const int s0   = (gw & 127) * 16;
  const int h    = (gw >> 7) & 7;
  const int b    = gw >> 10;
  const int r    = lane & 15, half = lane >> 4;

  const _Float16* qbase = qh + ((size_t)b * NHEADS + h) * SEQ * DK;
  const _Float16* kbase = kh + ((size_t)b * NHEADS + h) * SEQ * DK;
  const _Float16* vbase = vt + ((size_t)b * NHEADS + h) * DK * SEQ;

  // staging segment owned by this thread (256 segments of 16B per chunk)
  const int seg  = threadIdx.x;
  const int krow = seg >> 3, k8 = seg & 7;   // K chunk: 32 rows x 8 segs
  const int vrow = seg >> 2, v4 = seg & 3;   // V chunk: 64 rows x 4 segs
  auto stage = [&](int t0n, int bufn) {
    async_cp16(&kb[bufn][krow * KP + k8 * 8],
               kbase + (size_t)(t0n + krow) * DK + k8 * 8);
    async_cp16(&vb[bufn][vrow * VP + v4 * 8],
               vbase + (size_t)vrow * SEQ + t0n + v4 * 8);
  };

  // q^T B-fragments, loaded once per wave
  v16h qB[2];
#pragma unroll
  for (int ks = 0; ks < 2; ks++)
    qB[ks] = *(const v16h*)(qbase + (size_t)(s0 + r) * DK + ks * 32 + half * 16);

  v8f  o[4] = {v8f{}, v8f{}, v8f{}, v8f{}};
  float m = -INFINITY, l = 0.0f;
  const float rs = 0.125f;   // 1/sqrt(64)

  stage(0, 0);
  for (int it = 0; it < SEQ / 32; ++it) {
    const int t0  = it * 32;
    const int buf = it & 1;
    wait_async0();
    __syncthreads();            // chunk `it` visible; prev buffer free
    if (t0 + 32 < SEQ) stage(t0 + 32, buf ^ 1);

    // ---- scores^T tiles from LDS:  D[t][s] = sum_k K[t][k]*q[s][k]
    v8f stf[2] = {v8f{}, v8f{}};
#pragma unroll
    for (int tb = 0; tb < 2; tb++) {
#pragma unroll
      for (int ks = 0; ks < 2; ks++) {
        const _Float16* kp = &kb[buf][(tb * 16 + r) * KP + ks * 32 + half * 16];
        V16U ka;
        ka.h[0] = *(const v8h*)kp;
        ka.h[1] = *(const v8h*)(kp + 8);
        stf[tb] = wmma_f16(ka.v, qB[ks], stf[tb]);
      }
    }
    // ---- online softmax: column s = s0 + r lives in lanes r and r+16
    float cmax = -INFINITY;
#pragma unroll
    for (int tb = 0; tb < 2; tb++)
#pragma unroll
      for (int v = 0; v < 8; v++) cmax = fmaxf(cmax, stf[tb][v] * rs);
    cmax = fmaxf(cmax, __shfl_xor(cmax, 16, 32));
    float mnew  = fmaxf(m, cmax);
    float alpha = __expf(m - mnew);
    float p0[8], p1[8], csum = 0.0f;
#pragma unroll
    for (int v = 0; v < 8; v++) {
      p0[v] = __expf(stf[0][v] * rs - mnew);
      p1[v] = __expf(stf[1][v] * rs - mnew);
      csum += p0[v] + p1[v];
    }
    csum += __shfl_xor(csum, 16, 32);
    l = l * alpha + csum;
    m = mnew;
#pragma unroll
    for (int df = 0; df < 4; df++)
#pragma unroll
      for (int v = 0; v < 8; v++) o[df][v] *= alpha;

    // ---- P^T C-fragment -> B-fragment via half-swap shuffles (no LDS)
    v16h pB;
#pragma unroll
    for (int v = 0; v < 8; v++) {
      float a0 = p0[v], a1 = p1[v];
      float x0 = __shfl_xor(a0, 16, 32);
      float x1 = __shfl_xor(a1, 16, 32);
      pB[v]     = (_Float16)(half ? x1 : a0);
      pB[v + 8] = (_Float16)(half ? a1 : x0);
    }
    // ---- O^T += V^T x P^T  (V^T fragments from LDS)
#pragma unroll
    for (int df = 0; df < 4; df++) {
      const _Float16* vp = &vb[buf][(df * 16 + r) * VP + half * 16];
      V16U va;
      va.h[0] = *(const v8h*)vp;
      va.h[1] = *(const v8h*)(vp + 8);
      o[df] = wmma_f16(va.v, pB, o[df]);
    }
  }

  // normalize, merge heads: attnH[(b*S+s)*512 + h*64 + d], f16
  float inv_l = 1.0f / l;
  _Float16* obase = attnH + ((size_t)b * SEQ + s0 + r) * DMODEL + h * DK;
#pragma unroll
  for (int df = 0; df < 4; df++) {
    v8h pk;
#pragma unroll
    for (int v = 0; v < 8; v++) pk[v] = (_Float16)(o[df][v] * inv_l);
    *(v8h*)(obase + df * 16 + half * 8) = pk;
  }
}

// ---------------------------------------------------------------------------
// Kernel 4: output projection  out(f32) = attnH(f16)[M,512] @ Wo + bo
// Same LDS weight-slab scheme as proj_kernel.
// ---------------------------------------------------------------------------
__global__ void __launch_bounds__(256)
out_proj_kernel(const _Float16* __restrict__ Xh, const _Float16* __restrict__ Wt,
                const float* __restrict__ bias, float* __restrict__ out) {
  __shared__ __align__(16) _Float16 wl[64 * WP];

  const int lane = threadIdx.x & 31;
  const int wave = threadIdx.x >> 5;
  const int nb   = blockIdx.x & 7;
  const int n0   = nb * 64;
  const int m0   = ((blockIdx.x >> 3) * 8 + wave) * 16;
  const int r    = lane & 15, half = lane >> 4;

#pragma unroll
  for (int t = 0; t < 16; t++) {
    int seg = t * 256 + threadIdx.x;
    int row = seg >> 5, s8 = seg & 31;
    async_cp16(&wl[row * WP + s8 * 8], Wt + (size_t)(n0 + row) * DMODEL + s8 * 8);
  }
  wait_async0();
  __syncthreads();

  v8f acc[4] = {v8f{}, v8f{}, v8f{}, v8f{}};
  const _Float16* arow = Xh + (size_t)(m0 + r) * DMODEL;

  for (int k0 = 0; k0 < DMODEL; k0 += 32) {
    const _Float16* p0 = arow + k0 + half * 8;
    V16U a;
    a.h[0] = *(const v8h*)p0;
    a.h[1] = *(const v8h*)(p0 + 16);
#pragma unroll
    for (int nf = 0; nf < 4; nf++) {
      const _Float16* bp = &wl[(nf * 16 + r) * WP + k0 + half * 8];
      V16U bf;
      bf.h[0] = *(const v8h*)bp;
      bf.h[1] = *(const v8h*)(bp + 16);
      acc[nf] = wmma_f16(a.v, bf.v, acc[nf]);
    }
  }
#pragma unroll
  for (int nf = 0; nf < 4; nf++) {
    int   n  = n0 + nf * 16 + r;
    float bv = bias[n];
#pragma unroll
    for (int v = 0; v < 8; v++) {
      int mrow = m0 + half * 8 + v;
      out[(size_t)mrow * DMODEL + n] = acc[nf][v] + bv;
    }
  }
}

// ---------------------------------------------------------------------------
extern "C" void kernel_launch(void* const* d_in, const int* in_sizes, int n_in,
                              void* d_out, int out_size, void* d_ws, size_t ws_size,
                              hipStream_t stream) {
  const float* Q  = (const float*)d_in[0];
  const float* K  = (const float*)d_in[1];
  const float* V  = (const float*)d_in[2];
  const float* Wq = (const float*)d_in[3];
  const float* bq = (const float*)d_in[4];
  const float* Wk = (const float*)d_in[5];
  const float* bk = (const float*)d_in[6];
  const float* Wv = (const float*)d_in[7];
  const float* bv = (const float*)d_in[8];
  const float* Wo = (const float*)d_in[9];
  const float* bo = (const float*)d_in[10];

  const size_t WSZ = (size_t)4 * DMODEL * DMODEL;   // 4 transposed weights
  const size_t PSZ = (size_t)MTOT * DMODEL;         // each projection
  _Float16* Wt    = (_Float16*)d_ws;
  _Float16* qhp   = Wt + WSZ;
  _Float16* khp   = qhp + PSZ;
  _Float16* vtb   = khp + PSZ;
  _Float16* attnH = vtb + PSZ;

  transpose_weights_kernel<<<4096, 256, 0, stream>>>(Wq, Wk, Wv, Wo, Wt);

  proj_kernel<<<512, 256, 0, stream>>>(Q, Wt + 0 * DMODEL * DMODEL, bq, qhp, 0);
  proj_kernel<<<512, 256, 0, stream>>>(K, Wt + 1 * DMODEL * DMODEL, bk, khp, 0);
  proj_kernel<<<512, 256, 0, stream>>>(V, Wt + 2 * DMODEL * DMODEL, bv, vtb, 1);

  attn_kernel<<<512, 256, 0, stream>>>(qhp, khp, vtb, attnH);

  out_proj_kernel<<<512, 256, 0, stream>>>(attnH, Wt + 3 * DMODEL * DMODEL, bo,
                                           (float*)d_out);
  (void)in_sizes; (void)n_in; (void)out_size; (void)ws_size;
}
